// MultiheadScaledDotProductAttention_39676907886288
// MI455X (gfx1250) — matrix operand order repaired and checked by reading, not compiled, not run on hardware
//
#include <hip/hip_runtime.h>

typedef __attribute__((ext_vector_type(16))) __bf16 v16bf;
typedef __attribute__((ext_vector_type(8)))  __bf16 v8bf;
typedef __attribute__((ext_vector_type(8)))  float  v8f;

#define BQ  2
#define SEQ 2048
#define DIM 1024
#define NH  16
#define HD  64

union FragBF { v16bf v; v8bf h[2]; };

__device__ __forceinline__ v16bf load_frag_pair(const __bf16* p0, const __bf16* p1) {
  FragBF u;
  u.h[0] = *(const v8bf*)p0;
  u.h[1] = *(const v8bf*)p1;
  return u.v;
}

// CDNA5 async copy: global -> LDS, 16 bytes per lane, tracked by ASYNCcnt.
__device__ __forceinline__ void async_copy_b128(unsigned lds_addr, const void* gaddr) {
  asm volatile("global_load_async_to_lds_b128 %0, %1, off"
               :: "v"(lds_addr), "v"(gaddr) : "memory");
}
template<int N>
__device__ __forceinline__ void wait_asynccnt() {
  asm volatile("s_wait_asynccnt %0" :: "n"(N) : "memory");
}

// ---------------------------------------------------------------------------
// Tiled GEMM:  Out[m,n] = sum_k X[m,k] * W[n,k] + bias[n]
// Block tile 128x128, K-tile 32, 256 threads = 8 waves, each wave 64x32.
// Double-buffered LDS; global loads for tile k+1 issued before WMMA of tile k.
// IN_BF16: 0 = X fp32 (convert in staging), 1 = X bf16.
// OUT_MODE: 0 = bf16 [B,H,S,hd]; 1 = bf16 [B,H,hd,S]; 2 = fp32 [M,N].
// ---------------------------------------------------------------------------
template<int IN_BF16, int OUT_MODE>
__global__ __launch_bounds__(256)
void gemm_proj(const void* __restrict__ Xv_, const float* __restrict__ W,
               const float* __restrict__ bias, void* __restrict__ Outv,
               int M, int N, int K)
{
  __shared__ __bf16 As[2][128][40];   // row stride 80B (16B multiple)
  __shared__ __bf16 Bs[2][128][40];

  const int t    = threadIdx.x;
  const int lane = t & 31;
  const int wave = t >> 5;
  const int wm   = (wave >> 2) * 64;
  const int wn   = (wave & 3)  * 32;
  const int bm   = blockIdx.y * 128;
  const int bn   = blockIdx.x * 128;

  v8f zero = {};
  v8f acc[4][2];
#pragma unroll
  for (int i = 0; i < 4; ++i)
#pragma unroll
    for (int j = 0; j < 2; ++j) acc[i][j] = zero;

  const int mr = lane & 15;
  const int kb = (lane >> 4) * 8;      // A-frag K chunk base
  const int kh = (lane >> 4) * 8;      // B-frag K chunk base

  const int nkt = K >> 5;

  float4 aR[4], bR[4];
  v8bf   aRH[2];

  auto loadA = [&](int kt) {
    if (IN_BF16) {
      const __bf16* X = (const __bf16*)Xv_;
#pragma unroll
      for (int j = 0; j < 2; ++j) {
        int id = t + j * 256, row = id >> 2, kc = (id & 3) * 8;
        aRH[j] = *(const v8bf*)(X + (size_t)(bm + row) * K + kt * 32 + kc);
      }
    } else {
      const float* X = (const float*)Xv_;
#pragma unroll
      for (int j = 0; j < 4; ++j) {
        int id = t + j * 256, row = id >> 3, kc = (id & 7) * 4;
        aR[j] = *(const float4*)(X + (size_t)(bm + row) * K + kt * 32 + kc);
      }
    }
  };
  auto loadB = [&](int kt) {
#pragma unroll
    for (int j = 0; j < 4; ++j) {
      int id = t + j * 256, row = id >> 3, kc = (id & 7) * 4;
      bR[j] = *(const float4*)(W + (size_t)(bn + row) * K + kt * 32 + kc);
    }
  };
  auto storeA = [&](int buf) {
    if (IN_BF16) {
#pragma unroll
      for (int j = 0; j < 2; ++j) {
        int id = t + j * 256, row = id >> 2, kc = (id & 3) * 8;
        *(v8bf*)&As[buf][row][kc] = aRH[j];
      }
    } else {
#pragma unroll
      for (int j = 0; j < 4; ++j) {
        int id = t + j * 256, row = id >> 3, kc = (id & 7) * 4;
        As[buf][row][kc + 0] = (__bf16)aR[j].x;
        As[buf][row][kc + 1] = (__bf16)aR[j].y;
        As[buf][row][kc + 2] = (__bf16)aR[j].z;
        As[buf][row][kc + 3] = (__bf16)aR[j].w;
      }
    }
  };
  auto storeB = [&](int buf) {
#pragma unroll
    for (int j = 0; j < 4; ++j) {
      int id = t + j * 256, row = id >> 3, kc = (id & 7) * 4;
      Bs[buf][row][kc + 0] = (__bf16)bR[j].x;
      Bs[buf][row][kc + 1] = (__bf16)bR[j].y;
      Bs[buf][row][kc + 2] = (__bf16)bR[j].z;
      Bs[buf][row][kc + 3] = (__bf16)bR[j].w;
    }
  };

  // ---- pipeline prologue ----
  loadA(0); loadB(0);
  storeA(0); storeB(0);
  __syncthreads();

  for (int kt = 0; kt < nkt; ++kt) {
    const int  cur  = kt & 1;
    const bool more = (kt + 1 < nkt);
    if (more) { loadA(kt + 1); loadB(kt + 1); }       // overlap with WMMA below
    if (kt + 2 < nkt) {                               // global_prefetch_b8
      if (IN_BF16)
        __builtin_prefetch((const __bf16*)Xv_ + (size_t)(bm + (t >> 2)) * K + (kt + 2) * 32 + (t & 3) * 8, 0, 0);
      else
        __builtin_prefetch((const float*)Xv_ + (size_t)(bm + (t >> 3)) * K + (kt + 2) * 32 + (t & 7) * 4, 0, 0);
      __builtin_prefetch(W + (size_t)(bn + (t >> 3)) * K + (kt + 2) * 32 + (t & 7) * 4, 0, 0);
    }

    v16bf af[4], bfz[2];
#pragma unroll
    for (int mt = 0; mt < 4; ++mt) {
      const __bf16* p = &As[cur][wm + mt * 16 + mr][kb];
      af[mt] = load_frag_pair(p, p + 16);
    }
#pragma unroll
    for (int nt = 0; nt < 2; ++nt) {
      const __bf16* p = &Bs[cur][wn + nt * 16 + mr][kh];
      bfz[nt] = load_frag_pair(p, p + 16);
    }
#pragma unroll
    for (int mt = 0; mt < 4; ++mt)
#pragma unroll
      for (int nt = 0; nt < 2; ++nt)
        acc[mt][nt] = __builtin_amdgcn_wmma_f32_16x16x32_bf16(
            false, af[mt], false, bfz[nt], (short)0, acc[mt][nt], false, false);

    if (more) { storeA(1 - cur); storeB(1 - cur); }
    __syncthreads();
  }

  // ---- epilogue ----
  const int hlf = lane >> 4;
#pragma unroll
  for (int mt = 0; mt < 4; ++mt) {
#pragma unroll
    for (int nt = 0; nt < 2; ++nt) {
      int n = bn + wn + nt * 16 + mr;
      float bv = bias[n];
#pragma unroll
      for (int r = 0; r < 8; ++r) {
        int m = bm + wm + mt * 16 + r + 8 * hlf;
        float val = acc[mt][nt][r] + bv;
        if (OUT_MODE == 2) {
          ((float*)Outv)[(size_t)m * N + n] = val;
        } else {
          int bb = m / SEQ, ss = m % SEQ;
          int hh = n / HD,  dd = n % HD;
          size_t idx;
          if (OUT_MODE == 0) idx = ((size_t)((bb * NH + hh) * SEQ + ss)) * HD + dd;
          else               idx = ((size_t)((bb * NH + hh) * HD  + dd)) * SEQ + ss;
          ((__bf16*)Outv)[idx] = (__bf16)val;
        }
      }
    }
  }
}

// ---------------------------------------------------------------------------
// Flash attention: block = 128 q-rows (8 waves x 16), 64-key tiles.
// K/V tiles staged with GLOBAL_LOAD_ASYNC_TO_LDS_B128 + double buffering
// (ASYNCcnt split-wait: issue next tile, wait for the older 4 ops, barrier).
// ---------------------------------------------------------------------------
__global__ __launch_bounds__(256)
void attn_fa(const __bf16* __restrict__ Qh, const __bf16* __restrict__ Kh,
             const __bf16* __restrict__ Vt, const int* __restrict__ mask,
             __bf16* __restrict__ Ao)
{
  __shared__ __bf16 Ks[2][64][72];     // [key][hd], row stride 144B
  __shared__ __bf16 Vs[2][64][72];     // [hd][key]
  __shared__ __bf16 Ps[8][16][72];     // per-wave P tile (16 x 64)

  const int t = threadIdx.x, lane = t & 31, wave = t >> 5;
  const int bh = blockIdx.y, b = bh / NH, h = bh % NH;
  const int q0 = blockIdx.x * 128 + wave * 16;
  const size_t headQK = (size_t)(b * NH + h) * SEQ * HD;
  const size_t headV  = (size_t)(b * NH + h) * HD * SEQ;

  const int mr  = lane & 15;
  const int hlf = lane >> 4;

  // issue async staging of one 64-key tile into buffer `buf`
  auto issue_tile = [&](int buf, int kt0) {
#pragma unroll
    for (int j = 0; j < 2; ++j) {
      int id = t + j * 256, row = id >> 3, c = (id & 7) * 8;  // 16B chunks
      async_copy_b128((unsigned)(uintptr_t)&Ks[buf][row][c],
                      Kh + headQK + (size_t)(kt0 + row) * HD + c);
      async_copy_b128((unsigned)(uintptr_t)&Vs[buf][row][c],
                      Vt + headV + (size_t)row * SEQ + kt0 + c);
    }
  };

  // Q fragments: hd = 64 -> two K-steps of 32
  v16bf qa[2];
#pragma unroll
  for (int kk = 0; kk < 2; ++kk) {
    const __bf16* p = Qh + headQK + (size_t)(q0 + mr) * HD + kk * 32 + hlf * 8;
    qa[kk] = load_frag_pair(p, p + 16);
  }

  v8f zero = {};
  v8f o[4];
#pragma unroll
  for (int nt = 0; nt < 4; ++nt) o[nt] = zero;
  float mrow[8], lrow[8];
#pragma unroll
  for (int r = 0; r < 8; ++r) { mrow[r] = -3.0e38f; lrow[r] = 0.0f; }

  issue_tile(0, 0);                                  // prologue: 4 async ops

  for (int it = 0; it < SEQ / 64; ++it) {
    const int kt0 = it * 64;
    const int cur = it & 1;
    if (kt0 + 64 < SEQ) {
      issue_tile(1 - cur, kt0 + 64);                 // 4 newer async ops
      wait_asynccnt<4>();                            // current tile's 4 done
    } else {
      wait_asynccnt<0>();
    }
    __syncthreads();

    // ---- scores S = Q K^T (16 x 64) ----
    v8f s[4];
#pragma unroll
    for (int nt = 0; nt < 4; ++nt) {
      s[nt] = zero;
#pragma unroll
      for (int kk = 0; kk < 2; ++kk) {
        const __bf16* p = &Ks[cur][nt * 16 + mr][kk * 32 + hlf * 8];
        v16bf kfrag = load_frag_pair(p, p + 16);
        s[nt] = __builtin_amdgcn_wmma_f32_16x16x32_bf16(
            false, qa[kk], false, kfrag, (short)0, s[nt], false, false);
      }
    }

    // ---- scale + mask ----
#pragma unroll
    for (int nt = 0; nt < 4; ++nt) {
#pragma unroll
      for (int r = 0; r < 8; ++r) {
        int qrow = q0 + r + 8 * hlf;
        int key  = kt0 + nt * 16 + mr;
        float sv = s[nt][r] * 0.125f;   // 1/sqrt(64)
        int mv = mask[(size_t)b * SEQ * SEQ + (size_t)qrow * SEQ + key];
        s[nt][r] = (mv == 0) ? -1.0e9f : sv;
      }
    }

    // ---- online softmax (row reductions within 16-lane halves) ----
#pragma unroll
    for (int r = 0; r < 8; ++r) {
      float mx = fmaxf(fmaxf(s[0][r], s[1][r]), fmaxf(s[2][r], s[3][r]));
#pragma unroll
      for (int off = 1; off < 16; off <<= 1)
        mx = fmaxf(mx, __shfl_xor(mx, off, 32));
      float mnew = fmaxf(mrow[r], mx);
      float corr = __expf(mrow[r] - mnew);
      float psum = 0.0f;
#pragma unroll
      for (int nt = 0; nt < 4; ++nt) {
        float p = __expf(s[nt][r] - mnew);
        s[nt][r] = p;
        psum += p;
      }
#pragma unroll
      for (int off = 1; off < 16; off <<= 1)
        psum += __shfl_xor(psum, off, 32);
      lrow[r] = lrow[r] * corr + psum;
      mrow[r] = mnew;
#pragma unroll
      for (int nt = 0; nt < 4; ++nt) o[nt][r] *= corr;
    }

    // ---- redistribute P (C-layout) -> LDS -> A-layout ----
#pragma unroll
    for (int nt = 0; nt < 4; ++nt)
#pragma unroll
      for (int r = 0; r < 8; ++r)
        Ps[wave][r + 8 * hlf][nt * 16 + mr] = (__bf16)s[nt][r];

    // ---- O += P V (keys = reduction, 2 K-steps of 32) ----
#pragma unroll
    for (int kk = 0; kk < 2; ++kk) {
      const __bf16* pp = &Ps[wave][mr][kk * 32 + hlf * 8];
      v16bf pa = load_frag_pair(pp, pp + 16);
#pragma unroll
      for (int nt = 0; nt < 4; ++nt) {
        const __bf16* pv = &Vs[cur][nt * 16 + mr][kk * 32 + hlf * 8];
        v16bf vfrag = load_frag_pair(pv, pv + 16);
        o[nt] = __builtin_amdgcn_wmma_f32_16x16x32_bf16(
            false, pa, false, vfrag, (short)0, o[nt], false, false);
      }
    }
    __syncthreads();   // all waves done with `cur` before it is refilled
  }

  // ---- normalize and write merged-head output ----
#pragma unroll
  for (int nt = 0; nt < 4; ++nt) {
#pragma unroll
    for (int r = 0; r < 8; ++r) {
      int qrow = q0 + r + 8 * hlf;
      int d = nt * 16 + mr;
      float val = o[nt][r] / lrow[r];
      Ao[(size_t)(b * SEQ + qrow) * DIM + h * HD + d] = (__bf16)val;
    }
  }
}

// ---------------------------------------------------------------------------
extern "C" void kernel_launch(void* const* d_in, const int* in_sizes, int n_in,
                              void* d_out, int out_size, void* d_ws, size_t ws_size,
                              hipStream_t stream)
{
  const float* Xq  = (const float*)d_in[0];
  const float* Xk  = (const float*)d_in[1];
  const float* Xv  = (const float*)d_in[2];
  const int*   msk = (const int*)d_in[3];
  const float* Wq  = (const float*)d_in[4];
  const float* bq  = (const float*)d_in[5];
  const float* Wk  = (const float*)d_in[6];
  const float* bk  = (const float*)d_in[7];
  const float* Wv  = (const float*)d_in[8];
  const float* bv  = (const float*)d_in[9];
  const float* Wo  = (const float*)d_in[10];
  const float* bo  = (const float*)d_in[11];

  const size_t he = (size_t)BQ * NH * SEQ * HD;   // 4,194,304 elements
  __bf16* Qh = (__bf16*)d_ws;
  __bf16* Kh = Qh + he;
  __bf16* Vt = Kh + he;
  __bf16* Ao = Vt + he;

  dim3 blk(256, 1, 1);
  dim3 gg(DIM / 128, (BQ * SEQ) / 128, 1);        // (8, 32)

  gemm_proj<0, 0><<<gg, blk, 0, stream>>>(Xq, Wq, bq, Qh, BQ * SEQ, DIM, DIM);
  gemm_proj<0, 0><<<gg, blk, 0, stream>>>(Xk, Wk, bk, Kh, BQ * SEQ, DIM, DIM);
  gemm_proj<0, 1><<<gg, blk, 0, stream>>>(Xv, Wv, bv, Vt, BQ * SEQ, DIM, DIM);

  attn_fa<<<dim3(SEQ / 128, BQ * NH, 1), blk, 0, stream>>>(Qh, Kh, Vt, msk, Ao);

  gemm_proj<1, 2><<<gg, blk, 0, stream>>>(Ao, Wo, bo, d_out, BQ * SEQ, DIM, DIM);
}
